// BoneEncoder_14645838479863
// MI455X (gfx1250) — compile-verified, matrix operand order checked by reading, not run on hardware
//
#include <hip/hip_runtime.h>

#define NN 8192

typedef _Float16 v16h __attribute__((ext_vector_type(16)));
typedef float    v8f  __attribute__((ext_vector_type(8)));
typedef float    f4v  __attribute__((ext_vector_type(4)));

// ---------------------------------------------------------------------------
// Kernel 1: dinv[i] = rsqrt(deg_i), deg = sum of row i of A with diag forced 1.
// (A is symmetric so row-sum == the reference's column-sum; entries are exactly
// 0.0f / 1.0f so a plain float sum is an exact count.)
// ---------------------------------------------------------------------------
__global__ void gcn_dinv(const float* __restrict__ A, float* __restrict__ dinv) {
    int i = blockIdx.x;
    int t = threadIdx.x;
    const float* row = A + (size_t)i * NN;
    float s = 0.f;
    for (int j = t; j < NN; j += 256) {
        float a = row[j];
        s += (j == i) ? 1.0f : a;   // force self-loop weight 1
    }
    __shared__ float red[256];
    red[t] = s;
    __syncthreads();
    for (int off = 128; off > 0; off >>= 1) {
        if (t < off) red[t] += red[t + off];
        __syncthreads();
    }
    if (t == 0) dinv[i] = rsqrtf(red[0]);
}

// ---------------------------------------------------------------------------
// Kernel 2: TsT[h][j] = fp16( dinv[j] * (X @ W)[j][h] )   (transposed store so
// the aggregation kernel's WMMA B-operand loads are contiguous along K)
// ---------------------------------------------------------------------------
__global__ void gcn_feat(const float* __restrict__ X, const float* __restrict__ W,
                         const float* __restrict__ dinv, _Float16* __restrict__ TsT,
                         int IN, int HID) {
    int g = blockIdx.x * blockDim.x + threadIdx.x;   // HID*NN threads
    int h = g >> 13;            // feature column (uniform within a block)
    int j = g & (NN - 1);       // node index (consecutive -> coalesced store)
    const float* xr = X + (size_t)j * IN;
    float acc = 0.f;
    for (int k = 0; k < IN; ++k)
        acc += xr[k] * W[(size_t)k * HID + h];
    TsT[(size_t)h * NN + j] = (_Float16)(acc * dinv[j]);
}

// ---------------------------------------------------------------------------
// Kernel 3: out[i][col] = relu( dinv[i] * sum_j Afix[i][j]*Ts[j][col] + b[col] )
// One block = 16 output rows; wave w = 16 output cols (h0 = 16w).
// K-loop over all 8192 nodes in steps of 32 using v_wmma_f32_16x16x32_f16.
// A is {0,1} fp32 in memory -> fp16 convert is exact. Self-loop (+I) applied
// as an epilogue correction so the hot loop has no per-element compares.
// ---------------------------------------------------------------------------
__global__ void gcn_agg(const float* __restrict__ A, const _Float16* __restrict__ TsT,
                        const float* __restrict__ dinv, const float* __restrict__ bias,
                        float* __restrict__ out, int ncols) {
    int i0   = blockIdx.x << 4;          // 16-row stripe
    int wave = threadIdx.x >> 5;
    int lane = threadIdx.x & 31;
    int h0   = wave << 4;                // 16-col tile for this wave
    int l15  = lane & 15;
    int kb   = (lane < 16) ? 0 : 8;      // per-lane K sub-block per ISA A layout

    const float*    Arow = A   + (size_t)(i0 + l15) * NN;  // A fragment row (M = l15)
    const _Float16* Brow = TsT + (size_t)(h0 + l15) * NN;  // B fragment col (N = l15)

    v8f c = {0.f, 0.f, 0.f, 0.f, 0.f, 0.f, 0.f, 0.f};

    for (int j0 = 0; j0 < NN; j0 += 32) {
        int ja = j0 + kb;
        // A-operand: lane holds K = ja..ja+7 (elts 0..7) and ja+16..ja+23 (elts 8..15)
        f4v a0 = *(const f4v*)(Arow + ja);
        f4v a1 = *(const f4v*)(Arow + ja + 4);
        f4v a2 = *(const f4v*)(Arow + ja + 16);
        f4v a3 = *(const f4v*)(Arow + ja + 20);
        v16h av;
#pragma unroll
        for (int t = 0; t < 4; ++t) {
            av[t]      = (_Float16)a0[t];   // exact: values are 0.0 or 1.0
            av[4 + t]  = (_Float16)a1[t];
            av[8 + t]  = (_Float16)a2[t];
            av[12 + t] = (_Float16)a3[t];
        }
        // B-operand: same K sub-blocks, contiguous fp16 -> two b128 loads
        union { v16h h; f4v f[2]; } bu;
        bu.f[0] = *(const f4v*)(Brow + ja);
        bu.f[1] = *(const f4v*)(Brow + ja + 16);

        c = __builtin_amdgcn_wmma_f32_16x16x32_f16(
                /*neg_a=*/false, av, /*neg_b=*/false, bu.h,
                /*c_mod=*/(short)0, c, /*reuse_a=*/false, /*reuse_b=*/false);

        if (j0 + 256 < NN)
            __builtin_prefetch(Arow + ja + 256, 0, 0);   // global_prefetch_b8
    }

    // Epilogue per C/D layout: VGPR r, lane l -> row = r + 8*(l>=16), col = l&15
    int col   = h0 + l15;
    float bb  = bias[col];
    int rbase = (lane >> 4) << 3;
#pragma unroll
    for (int r = 0; r < 8; ++r) {
        int row = i0 + rbase + r;
        // self-loop correction: reference forces A[ii] = 1
        float diagA = A[(size_t)row * NN + row];
        float corr  = (diagA == 0.0f) ? (float)TsT[(size_t)col * NN + row] : 0.0f;
        float v = dinv[row] * (c[r] + corr) + bb;
        out[(size_t)row * ncols + col] = v > 0.0f ? v : 0.0f;
    }
}

// ---------------------------------------------------------------------------
// Host-side orchestration (graph-capture safe: launches only, on `stream`)
// ---------------------------------------------------------------------------
extern "C" void kernel_launch(void* const* d_in, const int* in_sizes, int n_in,
                              void* d_out, int out_size, void* d_ws, size_t ws_size,
                              hipStream_t stream) {
    const float* X  = (const float*)d_in[0];   // bone_features [8192,64]
    const float* A  = (const float*)d_in[1];   // bone_adj      [8192,8192]
    const float* W0 = (const float*)d_in[2];   // [64,128]
    const float* b0 = (const float*)d_in[3];   // [128]
    const float* W1 = (const float*)d_in[4];   // [128,128]
    const float* b1 = (const float*)d_in[5];   // [128]
    const float* W2 = (const float*)d_in[6];   // [128,64]
    const float* b2 = (const float*)d_in[7];   // [64]

    // Workspace layout (~10.1 MB total):
    //   [0,32KB)          dinv  (8192 f32)
    //   [32KB, 32KB+2MB)  TsT   (128*8192 f16, transposed scaled features)
    //   then two 4MB fp32 activation ping-pong buffers
    char* ws = (char*)d_ws;
    float*    dinv = (float*)ws;
    _Float16* TsT  = (_Float16*)(ws + (32u << 10));
    float*    XA   = (float*)(ws + (32u << 10) + (2u << 20));
    float*    XB   = XA + (size_t)NN * 128;

    gcn_dinv<<<NN, 256, 0, stream>>>(A, dinv);

    // Layer 0: 64 -> 128
    gcn_feat<<<(128 * NN) / 256, 256, 0, stream>>>(X, W0, dinv, TsT, 64, 128);
    gcn_agg <<<NN / 16, 256, 0, stream>>>(A, TsT, dinv, b0, XA, 128);

    // Layer 1: 128 -> 128
    gcn_feat<<<(128 * NN) / 256, 256, 0, stream>>>(XA, W1, dinv, TsT, 128, 128);
    gcn_agg <<<NN / 16, 256, 0, stream>>>(A, TsT, dinv, b1, XB, 128);

    // Layer 2: 128 -> 64 (4 waves per block, one per 16-col tile)
    gcn_feat<<<(64 * NN) / 256, 256, 0, stream>>>(XB, W2, dinv, TsT, 128, 64);
    gcn_agg <<<NN / 16, 128, 0, stream>>>(A, TsT, dinv, b2, (float*)d_out, 64);
}
